// MoRALayer_52055003627791
// MI455X (gfx1250) — matrix operand order repaired and checked by reading, not compiled, not run on hardware
//
#include <hip/hip_runtime.h>
#include <hip/hip_bf16.h>

// ---------------------------------------------------------------------------
// MoRA layer for MI455X (gfx1250, wave32, WMMA + TDM).
// GEMMs: v_wmma_f32_16x16x32_bf16, fp32 accumulation.
// Global->LDS staging: Tensor Data Mover (tensor_load_to_lds), double-buffered.
// ---------------------------------------------------------------------------

typedef __bf16 bf16_t;
typedef __attribute__((ext_vector_type(8)))  __bf16 v8bf;
typedef __attribute__((ext_vector_type(16))) __bf16 v16bf;
typedef __attribute__((ext_vector_type(8)))  float  v8f;

typedef unsigned int u32x4 __attribute__((ext_vector_type(4)));
typedef int          i32x4 __attribute__((ext_vector_type(4)));
typedef int          i32x8 __attribute__((ext_vector_type(8)));

#define B_ROWS 16384
#define DFEAT  768
#define HGATE  3072          // 4*H
#define T_STEPS 10

#define BM 128
#define BN 128
#define BK 32
#define LDSS 40              // LDS row stride in bf16 (80 B = 64 B data + 16 B pad)

#if __has_builtin(__builtin_amdgcn_tensor_load_to_lds) && \
    __has_builtin(__builtin_amdgcn_s_wait_tensorcnt)
#define USE_TDM 1
#else
#define USE_TDM 0
#endif

__device__ __forceinline__ float sigmoid_f(float x) {
    return 1.0f / (1.0f + __expf(-x));
}

#if USE_TDM
typedef __attribute__((address_space(3))) void lds_void;

__device__ __forceinline__ unsigned lds_addr_of(const void* p) {
    return (unsigned)(unsigned long long)(lds_void*)p;
}

// Issue one TDM load of a 128-row x 32-col bf16 tile into LDS with
// 16 B padding after every 64 B row (matches LDSS). Wave-level op.
// This toolchain's builtin takes 6 args (g0 u32x4, g1 i32x8, i32x4, i32x4,
// i32x8, i32 cpol); the 2-D descriptor lives in groups 0/1, rest zero.
__device__ __forceinline__ void tdm_load_tile(const bf16_t* gsrc,
                                              unsigned lds_byte_addr,
                                              int row_stride_elems) {
    const unsigned long long ga = (unsigned long long)gsrc;
    u32x4 g0;
    g0[0] = 1u;                                                // count=1, user D#
    g0[1] = lds_byte_addr;                                     // lds_addr
    g0[2] = (unsigned)(ga & 0xFFFFFFFFu);                      // global_addr[31:0]
    g0[3] = (unsigned)((ga >> 32) & 0x01FFFFFFu) | (2u << 30); // [56:32] | type=2

    // tensor dims set huge: tiles are always fully interior -> no OOB clipping.
    const unsigned td0 = 0x7FFFFFFFu, td1 = 0x7FFFFFFFu;
    i32x8 g1;
    g1[0] = (int)((1u << 16)        // data_size = 2 bytes
                | (1u << 20)        // pad_enable
                | (3u << 22)        // pad_interval: 16 DWORDs (64 B)
                | (3u << 25));      // pad_amount:    4 DWORDs (16 B)
    g1[1] = (int)((td0 & 0xFFFFu) << 16);                          // tensor_dim0 lo
    g1[2] = (int)(((td0 >> 16) & 0xFFFFu) | ((td1 & 0xFFFFu) << 16));
    g1[3] = (int)(((td1 >> 16) & 0xFFFFu) | (32u << 16));          // tile_dim0 = 32
    g1[4] = (int)128u;                                             // tile_dim1 = 128
    g1[5] = (int)(unsigned)row_stride_elems;                       // dim0_stride lo
    g1[6] = 0;                                                     // dim0_stride hi
    g1[7] = 0;                                                     // dim1_stride

    const i32x4 z4 = {0, 0, 0, 0};
    const i32x8 z8 = {0, 0, 0, 0, 0, 0, 0, 0};
    __builtin_amdgcn_tensor_load_to_lds(g0, g1, z4, z4, z8, 0);
}
#endif

// ---------------------------------------------------------------------------
// Tiled bf16 WMMA GEMM:  C[M,N] = A[M,K](lda) * Bw[N,K]^T  (+ epilogue fusion)
// mode 0: outf = acc + bias[col]
// mode 1: outf = acc + addend[row*ldout+col]
// mode 2: outbf = bf16(relu(acc + bias[col]))
// mode 3: t = rowscale[row*rs_stride] * (acc + bias[col]);
//         outf = t (beta_acc==0) or outf += t (beta_acc==1)
// All problem dims are multiples of the tile sizes -> no bounds checks.
// ---------------------------------------------------------------------------
__global__ __launch_bounds__(256)
void gemm_bf16_wmma(const bf16_t* __restrict__ A, const bf16_t* __restrict__ Bw,
                    int K, int lda,
                    float* __restrict__ outf, bf16_t* __restrict__ outbf,
                    const float* __restrict__ bias,
                    const float* __restrict__ addend,
                    const float* __restrict__ rowscale, int rs_stride,
                    int ldout, int mode, int beta_acc) {
    __shared__ bf16_t As[2][BM * LDSS];
    __shared__ bf16_t Bs[2][BN * LDSS];

    const int tid  = threadIdx.x;
    const int wid  = tid >> 5;
    const int lane = tid & 31;
    const int wm   = wid & 3;          // 4 waves along M (32 rows each)
    const int wn   = wid >> 2;         // 2 waves along N (64 cols each)
    const int blockM = blockIdx.y * BM;
    const int blockN = blockIdx.x * BN;

    const int l16     = lane & 15;
    const int halfsel = lane >> 4;     // 0: lanes 0-15, 1: lanes 16-31
    const int kfragA  = halfsel * 8;   // A: K[kf..kf+8) and K[kf+16..kf+24)

    const bf16_t* Ablk = A + (size_t)blockM * lda;
    const bf16_t* Bblk = Bw + (size_t)blockN * K;

    v8f acc[2][4];
#pragma unroll
    for (int mt = 0; mt < 2; ++mt)
#pragma unroll
        for (int nt = 0; nt < 4; ++nt)
            acc[mt][nt] = (v8f){};

#if USE_TDM
    const bool issuer = (wid == 0);
    if (issuer) {
        tdm_load_tile(Ablk, lds_addr_of(&As[0][0]), lda);
        tdm_load_tile(Bblk, lds_addr_of(&Bs[0][0]), K);
    }
#else
    // prologue: stage tile 0 manually (batched loads, then stores)
    {
        uint4 ra[2], rb[2];
#pragma unroll
        for (int t = 0; t < 2; ++t) {
            const int c = tid + t * 256;
            const int row = c >> 2, seg = c & 3;
            ra[t] = *reinterpret_cast<const uint4*>(Ablk + (size_t)row * lda + seg * 8);
            rb[t] = *reinterpret_cast<const uint4*>(Bblk + (size_t)row * K + seg * 8);
        }
#pragma unroll
        for (int t = 0; t < 2; ++t) {
            const int c = tid + t * 256;
            const int row = c >> 2, seg = c & 3;
            *reinterpret_cast<uint4*>(&As[0][row * LDSS + seg * 8]) = ra[t];
            *reinterpret_cast<uint4*>(&Bs[0][row * LDSS + seg * 8]) = rb[t];
        }
    }
#endif

    int buf = 0;
    for (int k0 = 0; k0 < K; k0 += BK, buf ^= 1) {
#if USE_TDM
        __builtin_amdgcn_s_wait_tensorcnt((short)0);
#endif
        __syncthreads();

        const int kn = k0 + BK;
#if USE_TDM
        if (issuer && kn < K) {
            tdm_load_tile(Ablk + kn, lds_addr_of(&As[buf ^ 1][0]), lda);
            tdm_load_tile(Bblk + kn, lds_addr_of(&Bs[buf ^ 1][0]), K);
        }
#else
        uint4 ra[2], rb[2];
        if (kn < K) {
#pragma unroll
            for (int t = 0; t < 2; ++t) {
                const int c = tid + t * 256;
                const int row = c >> 2, seg = c & 3;
                ra[t] = *reinterpret_cast<const uint4*>(Ablk + (size_t)row * lda + kn + seg * 8);
                rb[t] = *reinterpret_cast<const uint4*>(Bblk + (size_t)row * K + kn + seg * 8);
            }
        }
#endif

        // ---- fragments per ISA 7.12.2 layouts ----
        v16bf afrag[2], bfrag[4];
#pragma unroll
        for (int mt = 0; mt < 2; ++mt) {
            const int row = wm * 32 + mt * 16 + l16;
            v8bf lo = *reinterpret_cast<const v8bf*>(&As[buf][row * LDSS + kfragA]);
            v8bf hi = *reinterpret_cast<const v8bf*>(&As[buf][row * LDSS + kfragA + 16]);
            afrag[mt] = __builtin_shufflevector(lo, hi,
                0,1,2,3,4,5,6,7,8,9,10,11,12,13,14,15);
        }
#pragma unroll
        for (int nt = 0; nt < 4; ++nt) {
            const int col = wn * 64 + nt * 16 + l16;
            const bf16_t* base = &Bs[buf][col * LDSS + halfsel * 8];
            v8bf lo = *reinterpret_cast<const v8bf*>(base);
            v8bf hi = *reinterpret_cast<const v8bf*>(base + 16);
            bfrag[nt] = __builtin_shufflevector(lo, hi,
                0,1,2,3,4,5,6,7,8,9,10,11,12,13,14,15);
        }

        // ---- 8 WMMAs per K-step per wave ----
#pragma unroll
        for (int mt = 0; mt < 2; ++mt)
#pragma unroll
            for (int nt = 0; nt < 4; ++nt)
                acc[mt][nt] = __builtin_amdgcn_wmma_f32_16x16x32_bf16(
                    false, afrag[mt], false, bfrag[nt],
                    (short)0, acc[mt][nt], false, false);

#if !USE_TDM
        if (kn < K) {
#pragma unroll
            for (int t = 0; t < 2; ++t) {
                const int c = tid + t * 256;
                const int row = c >> 2, seg = c & 3;
                *reinterpret_cast<uint4*>(&As[buf ^ 1][row * LDSS + seg * 8]) = ra[t];
                *reinterpret_cast<uint4*>(&Bs[buf ^ 1][row * LDSS + seg * 8]) = rb[t];
            }
        }
#endif
    }

    // ---- epilogue: C/D layout = VGPR r -> M=r+8*halfsel, N=l16 ----
#pragma unroll
    for (int mt = 0; mt < 2; ++mt) {
#pragma unroll
        for (int nt = 0; nt < 4; ++nt) {
            const int col = blockN + wn * 64 + nt * 16 + l16;
#pragma unroll
            for (int r = 0; r < 8; ++r) {
                const size_t row = (size_t)blockM + wm * 32 + mt * 16 + r + halfsel * 8;
                const float v = acc[mt][nt][r];
                const size_t oi = row * (size_t)ldout + col;
                if (mode == 0) {
                    outf[oi] = v + bias[col];
                } else if (mode == 1) {
                    outf[oi] = v + addend[oi];
                } else if (mode == 2) {
                    float t = v + bias[col];
                    t = t > 0.0f ? t : 0.0f;
                    outbf[oi] = (bf16_t)t;
                } else { // mode 3
                    const float t = rowscale[row * rs_stride] * (v + bias[col]);
                    if (beta_acc) outf[oi] += t; else outf[oi] = t;
                }
            }
        }
    }
}

// ---------------------------------------------------------------------------
// LSTM pointwise + halting head (one block per batch row).
// ---------------------------------------------------------------------------
__global__ __launch_bounds__(256)
void lstm_pointwise(const float* __restrict__ gates,   // [B, 4H] (i,f,g,o)
                    float* __restrict__ c, float* __restrict__ h,
                    bf16_t* __restrict__ hbf, float* __restrict__ hsum,
                    float* __restrict__ hp, float* __restrict__ rem,
                    const float* __restrict__ Whalt, const float* __restrict__ bhalt) {
    __shared__ float red[256];
    const int row = blockIdx.x;
    const float* g = gates + (size_t)row * HGATE;
    float partial = 0.0f;
#pragma unroll
    for (int j = threadIdx.x; j < DFEAT; j += 256) {
        const float gi = g[j], gf = g[DFEAT + j], gg = g[2 * DFEAT + j], go = g[3 * DFEAT + j];
        const size_t idx = (size_t)row * DFEAT + j;
        const float cn = sigmoid_f(gf) * c[idx] + sigmoid_f(gi) * tanhf(gg);
        const float hn = sigmoid_f(go) * tanhf(cn);
        c[idx] = cn;
        h[idx] = hn;
        hbf[idx] = (bf16_t)hn;
        hsum[idx] += hn;
        partial += hn * Whalt[j];
    }
    red[threadIdx.x] = partial;
    __syncthreads();
#pragma unroll
    for (int s = 128; s > 0; s >>= 1) {
        if (threadIdx.x < s) red[threadIdx.x] += red[threadIdx.x + s];
        __syncthreads();
    }
    if (threadIdx.x == 0) {
        const float y   = sigmoid_f(red[0] + bhalt[0]);
        const float hpv = hp[row];
        const float hpn = hpv + y * (1.0f - hpv);
        hp[row]  = hpn;
        rem[row] += 1.0f - hpn;
    }
}

// avg = rem * hsum / T  -> both f32 and bf16 mirrors
__global__ void act_avg(const float* __restrict__ hsum, const float* __restrict__ rem,
                        float* __restrict__ h, bf16_t* __restrict__ hbf) {
    const size_t i = (size_t)blockIdx.x * 256 + threadIdx.x;
    if (i >= (size_t)B_ROWS * DFEAT) return;
    const int row = (int)(i / DFEAT);
    const float v = rem[row] * hsum[i] * (1.0f / (float)T_STEPS);
    h[i] = v;
    hbf[i] = (bf16_t)v;
}

// ---------------------------------------------------------------------------
// Top-2-of-4 routing: dense weight table wfull[b][e] (0 for unselected).
// ---------------------------------------------------------------------------
__global__ __launch_bounds__(256)
void route_top2(const float* __restrict__ h, const float* __restrict__ gW,
                const float* __restrict__ gb, float* __restrict__ wfull) {
    __shared__ float red[4 * 256];
    const int row = blockIdx.x;
    float p[4] = {0.f, 0.f, 0.f, 0.f};
#pragma unroll
    for (int j = threadIdx.x; j < DFEAT; j += 256) {
        const float x = h[(size_t)row * DFEAT + j];
#pragma unroll
        for (int e = 0; e < 4; ++e) p[e] += x * gW[e * DFEAT + j];
    }
#pragma unroll
    for (int e = 0; e < 4; ++e) red[e * 256 + threadIdx.x] = p[e];
    __syncthreads();
#pragma unroll
    for (int s = 128; s > 0; s >>= 1) {
        if (threadIdx.x < s)
#pragma unroll
            for (int e = 0; e < 4; ++e)
                red[e * 256 + threadIdx.x] += red[e * 256 + threadIdx.x + s];
        __syncthreads();
    }
    if (threadIdx.x == 0) {
        float l[4];
#pragma unroll
        for (int e = 0; e < 4; ++e) l[e] = red[e * 256] + gb[e];
        int i0 = 0;
#pragma unroll
        for (int e = 1; e < 4; ++e) if (l[e] > l[i0]) i0 = e;
        int i1 = -1;
#pragma unroll
        for (int e = 0; e < 4; ++e)
            if (e != i0 && (i1 < 0 || l[e] > l[i1])) i1 = e;
        const float e1 = __expf(l[i1] - l[i0]);
        const float inv = 1.0f / (1.0f + e1);
#pragma unroll
        for (int e = 0; e < 4; ++e)
            wfull[(size_t)row * 4 + e] = (e == i0) ? inv : (e == i1 ? e1 * inv : 0.0f);
    }
}

// copy f32 buffer into f32 + bf16 mirrors (new h after a MoE level)
__global__ void sync_h(const float* __restrict__ src, float* __restrict__ h,
                       bf16_t* __restrict__ hbf) {
    const size_t i = (size_t)blockIdx.x * 256 + threadIdx.x;
    if (i >= (size_t)B_ROWS * DFEAT) return;
    const float v = src[i];
    h[i] = v;
    hbf[i] = (bf16_t)v;
}

__global__ void convert_bf16(const float* __restrict__ in, bf16_t* __restrict__ out, int n) {
    const int i = blockIdx.x * 256 + threadIdx.x;
    if (i < n) out[i] = (bf16_t)in[i];
}

__global__ void add_bias_pair(const float* __restrict__ a, const float* __restrict__ b,
                              float* __restrict__ out, int n) {
    const int i = blockIdx.x * 256 + threadIdx.x;
    if (i < n) out[i] = a[i] + b[i];
}

// ---------------------------------------------------------------------------
// LayerNorm over last dim (one block per row).
// ---------------------------------------------------------------------------
__global__ __launch_bounds__(256)
void layernorm(const float* __restrict__ h, const float* __restrict__ gamma,
               const float* __restrict__ beta, float* __restrict__ out) {
    __shared__ float red[256];
    __shared__ float mean_s, rstd_s;
    const int row = blockIdx.x;
    const float* hr = h + (size_t)row * DFEAT;
    float s = 0.0f;
    for (int j = threadIdx.x; j < DFEAT; j += 256) s += hr[j];
    red[threadIdx.x] = s;
    __syncthreads();
#pragma unroll
    for (int st = 128; st > 0; st >>= 1) {
        if (threadIdx.x < st) red[threadIdx.x] += red[threadIdx.x + st];
        __syncthreads();
    }
    if (threadIdx.x == 0) mean_s = red[0] / (float)DFEAT;
    __syncthreads();
    const float mu = mean_s;
    float vs = 0.0f;
    for (int j = threadIdx.x; j < DFEAT; j += 256) {
        const float d = hr[j] - mu;
        vs += d * d;
    }
    red[threadIdx.x] = vs;
    __syncthreads();
#pragma unroll
    for (int st = 128; st > 0; st >>= 1) {
        if (threadIdx.x < st) red[threadIdx.x] += red[threadIdx.x + st];
        __syncthreads();
    }
    if (threadIdx.x == 0) rstd_s = rsqrtf(red[0] / (float)DFEAT + 1e-5f);
    __syncthreads();
    const float rstd = rstd_s;
    for (int j = threadIdx.x; j < DFEAT; j += 256)
        out[(size_t)row * DFEAT + j] = (hr[j] - mu) * rstd * gamma[j] + beta[j];
}

// ---------------------------------------------------------------------------
// Host orchestration
// ---------------------------------------------------------------------------
extern "C" void kernel_launch(void* const* d_in, const int* in_sizes, int n_in,
                              void* d_out, int out_size, void* d_ws, size_t ws_size,
                              hipStream_t stream) {
    const float* x     = (const float*)d_in[0];
    const float* Wi    = (const float*)d_in[1];
    const float* Wh    = (const float*)d_in[2];
    const float* bi    = (const float*)d_in[3];
    const float* bh    = (const float*)d_in[4];
    const float* Whalt = (const float*)d_in[5];
    const float* bhalt = (const float*)d_in[6];
    const float* gateW = (const float*)d_in[7];
    const float* gateb = (const float*)d_in[8];
    const float* W1    = (const float*)d_in[9];
    const float* b1    = (const float*)d_in[10];
    const float* W2    = (const float*)d_in[11];
    const float* b2    = (const float*)d_in[12];
    const float* gamma = (const float*)d_in[13];
    const float* betap = (const float*)d_in[14];
    float* out = (float*)d_out;

    char* p = (char*)d_ws;
    auto alloc = [&](size_t bytes) -> char* {
        char* r = p;
        p += (bytes + 255) & ~(size_t)255;
        return r;
    };

    bf16_t* Wi_bf   = (bf16_t*)alloc((size_t)HGATE * DFEAT * 2);
    bf16_t* Wh_bf   = (bf16_t*)alloc((size_t)HGATE * DFEAT * 2);
    bf16_t* W1_bf   = (bf16_t*)alloc((size_t)2 * HGATE * DFEAT * 2);
    bf16_t* W2_bf   = (bf16_t*)alloc((size_t)2 * HGATE * DFEAT * 2);
    bf16_t* x_bf    = (bf16_t*)alloc((size_t)B_ROWS * DFEAT * 2);
    float*  bias_ih = (float*) alloc((size_t)HGATE * 4);
    float*  h_f     = (float*) alloc((size_t)B_ROWS * DFEAT * 4);
    float*  c_f     = (float*) alloc((size_t)B_ROWS * DFEAT * 4);
    float*  hsum    = (float*) alloc((size_t)B_ROWS * DFEAT * 4);
    bf16_t* h_bf    = (bf16_t*)alloc((size_t)B_ROWS * DFEAT * 2);
    float*  hp      = (float*) alloc((size_t)B_ROWS * 4);
    float*  rem     = (float*) alloc((size_t)B_ROWS * 4);
    float*  wfull   = (float*) alloc((size_t)B_ROWS * 4 * 4);
    float*  xWi     = (float*) alloc((size_t)B_ROWS * HGATE * 4); // aliased as moe_out
    float*  gates   = (float*) alloc((size_t)B_ROWS * HGATE * 4); // aliased as hmid_bf
    float*  moe_out = xWi;
    bf16_t* hmid_bf = (bf16_t*)gates;

    const int nW  = HGATE * DFEAT;            // 2,359,296
    const int nW4 = 2 * HGATE * DFEAT;        // 4,718,592
    const int nX  = B_ROWS * DFEAT;           // 12,582,912

    // --- weight / activation conversion to bf16 ---
    convert_bf16<<<(nW  + 255) / 256, 256, 0, stream>>>(Wi, Wi_bf, nW);
    convert_bf16<<<(nW  + 255) / 256, 256, 0, stream>>>(Wh, Wh_bf, nW);
    convert_bf16<<<(nW4 + 255) / 256, 256, 0, stream>>>(W1, W1_bf, nW4);
    convert_bf16<<<(nW4 + 255) / 256, 256, 0, stream>>>(W2, W2_bf, nW4);
    convert_bf16<<<(nX  + 255) / 256, 256, 0, stream>>>(x,  x_bf,  nX);
    add_bias_pair<<<(HGATE + 255) / 256, 256, 0, stream>>>(bi, bh, bias_ih, HGATE);

    // --- zero recurrent state ---
    (void)hipMemsetAsync(c_f,  0, (size_t)nX * 4, stream);
    (void)hipMemsetAsync(hsum, 0, (size_t)nX * 4, stream);
    (void)hipMemsetAsync(h_bf, 0, (size_t)nX * 2, stream);
    (void)hipMemsetAsync(hp,   0, (size_t)B_ROWS * 4, stream);
    (void)hipMemsetAsync(rem,  0, (size_t)B_ROWS * 4, stream);

    dim3 gridWide(HGATE / BN, B_ROWS / BM);   // 24 x 128
    dim3 gridNarrow(DFEAT / BN, B_ROWS / BM); //  6 x 128

    // --- xWi = x @ Wi^T + (bi + bh)  (constant across steps) ---
    gemm_bf16_wmma<<<gridWide, 256, 0, stream>>>(
        x_bf, Wi_bf, DFEAT, DFEAT, xWi, nullptr, bias_ih,
        nullptr, nullptr, 0, HGATE, 0, 0);

    // --- T LSTM steps with halting head ---
    for (int t = 0; t < T_STEPS; ++t) {
        gemm_bf16_wmma<<<gridWide, 256, 0, stream>>>(
            h_bf, Wh_bf, DFEAT, DFEAT, gates, nullptr, nullptr,
            xWi, nullptr, 0, HGATE, 1, 0);
        lstm_pointwise<<<B_ROWS, 256, 0, stream>>>(
            gates, c_f, h_f, h_bf, hsum, hp, rem, Whalt, bhalt);
    }
    act_avg<<<(nX + 255) / 256, 256, 0, stream>>>(hsum, rem, h_f, h_bf);

    // --- two hierarchical sparse-MoE levels (dense compute, routed combine) ---
    for (int l = 0; l < 2; ++l) {
        route_top2<<<B_ROWS, 256, 0, stream>>>(
            h_f, gateW + (size_t)l * 4 * DFEAT, gateb + (size_t)l * 4, wfull);
        // hmid[:, e*H + j] = relu(h @ W1[l,e]^T + b1[l,e])  for all experts at once
        gemm_bf16_wmma<<<gridWide, 256, 0, stream>>>(
            h_bf, W1_bf + (size_t)l * HGATE * DFEAT, DFEAT, DFEAT,
            nullptr, hmid_bf, b1 + (size_t)l * HGATE,
            nullptr, nullptr, 0, HGATE, 2, 0);
        // weighted expert outputs accumulated into moe_out
        for (int e = 0; e < 4; ++e) {
            gemm_bf16_wmma<<<gridNarrow, 256, 0, stream>>>(
                hmid_bf + (size_t)e * DFEAT,
                W2_bf + ((size_t)l * 4 + e) * DFEAT * DFEAT,
                DFEAT, HGATE,
                moe_out, nullptr, b2 + ((size_t)l * 4 + e) * DFEAT,
                nullptr, wfull + e, 4, DFEAT, 3, e > 0 ? 1 : 0);
        }
        sync_h<<<(nX + 255) / 256, 256, 0, stream>>>(moe_out, h_f, h_bf);
    }

    // --- final LayerNorm ---
    layernorm<<<B_ROWS, 256, 0, stream>>>(h_f, gamma, betap, out);
}
